// WindowAttention_16209206575622
// MI455X (gfx1250) — matrix operand order repaired and checked by reading, not compiled, not run on hardware
//
#include <hip/hip_runtime.h>

// Swin window attention, fused single kernel for gfx1250 (MI455X).
// Memory-bound (~0.82 GB HBM traffic, ~35us floor @23.3TB/s): one coalesced
// b128 pass over qkv, f16 WMMA (v_wmma_f32_16x16x32_f16) for both matmuls,
// all intermediates in LDS. One workgroup per window, 8 wave32 = 8 heads.
// LDS (223 KB) limits to 1 WG/WGP, so VGPRs are free -> hoist fragments for
// back-to-back WMMA issue without ds-wait stalls.

#define L      49
#define LP     64
#define NH     8
#define DK     32
#define NWIN   64
#define CH3    768
#define QKV_PER_WIN (CH3 * L)      // 37632
#define SCALE  0.17677669529663689f

typedef _Float16 v2h  __attribute__((ext_vector_type(2)));
typedef _Float16 v8h  __attribute__((ext_vector_type(8)));
typedef _Float16 v16h __attribute__((ext_vector_type(16)));
typedef float    v2f  __attribute__((ext_vector_type(2)));
typedef float    v4f  __attribute__((ext_vector_type(4)));
typedef float    v8f  __attribute__((ext_vector_type(8)));

// LDS byte layout (dynamic shared, total 223296 B < 320 KB/WGP):
//  [0,      32768)  qT  f16 [8][64][32]  -- aliased by P f16 [8][64][64] after matmul1
//  [32768,  65536)  kT  f16 [8][64][32]
//  [65536,  98304)  vS  f16 [8][32][64]
//  [98304, 198656)  S   f32 [8][49][64]
//  [198656,208272)  mask f32 [49*49]
//  [208272,217888)  idx  i32 [49*49]
//  [217888,223296)  bias f32 [169*8]
#define OFF_KT     32768
#define OFF_V      65536
#define OFF_S      98304
#define OFF_MASK   198656
#define OFF_IDX    208272
#define OFF_BIAS   217888
#define SMEM_BYTES 223296

// A-matrix 16x32 f16 fragment: lanes 0-15 hold K kb+0..15 contiguous,
// lanes 16-31 hold K kb+16..31, row M = lane&15 (matches load_b_frag).
__device__ __forceinline__ v16h load_a_frag(const _Float16* rowBase, int kBase, int laneHi) {
    return *(const v16h*)(rowBase + kBase + (laneHi ? 16 : 0));
}

// B-matrix 32x16 f16 fragment: lane n = col, lanes 0-15 hold K kb+0..15
// contiguous, lanes 16-31 hold K kb+16..31 (per sparse-B layout pattern).
__device__ __forceinline__ v16h load_b_frag(const _Float16* colBase, int kBase, int laneHi) {
    return *(const v16h*)(colBase + kBase + (laneHi ? 16 : 0));
}

// Softmax over the 8 heads for one (t,s) column (reference softmaxes axis=1).
__device__ __forceinline__ void head_softmax(float x[NH], const float* bt, float madd) {
    float mx = -1e30f;
    #pragma unroll
    for (int h = 0; h < NH; ++h) { x[h] += bt[h] + madd; mx = fmaxf(mx, x[h]); }
    float sum = 0.0f;
    #pragma unroll
    for (int h = 0; h < NH; ++h) { x[h] = __expf(x[h] - mx); sum += x[h]; }
    const float inv = 1.0f / sum;
    #pragma unroll
    for (int h = 0; h < NH; ++h) x[h] *= inv;
}

__global__ __launch_bounds__(256) void swin_window_attn_kernel(
    const float* __restrict__ qkv,        // (B, 768, 49)
    const float* __restrict__ mask,       // (64, 49, 49)
    const float* __restrict__ bias_table, // (169, 8)
    const int*   __restrict__ rel_idx,    // (49, 49)
    float*       __restrict__ out)        // (B, 256, 49)
{
    extern __shared__ char smem[];
    _Float16* qT = (_Float16*)smem;                 // [8][64][32] row t, col c (scaled q)
    _Float16* kT = (_Float16*)(smem + OFF_KT);      // [8][64][32] row s, col c
    _Float16* vS = (_Float16*)(smem + OFF_V);       // [8][32][64] row c, col s
    float*    S  = (float*)(smem + OFF_S);          // [8][49][64] row t, col s
    float*    mS = (float*)(smem + OFF_MASK);
    int*      iS = (int*)(smem + OFF_IDX);
    float*    bS = (float*)(smem + OFF_BIAS);
    _Float16* P  = (_Float16*)smem;                 // [8][64][64], aliases qT+kT

    const int tid    = threadIdx.x;
    const int b      = blockIdx.x;
    const int w      = b & (NWIN - 1);
    const int lane   = tid & 31;
    const int wv     = tid >> 5;                    // wave id == head
    const int lane16 = lane & 15;
    const int laneHi = (lane >> 4) & 1;

    // ---------- phase 1: stage + convert to LDS (coalesced b128 reads) ----------
    {   // zero v pad columns s=49..63 (disjoint from converted region; NaN-proof P*V)
        const int c = tid & 31, h = tid >> 5;
        _Float16* vr = vS + (h * DK + c) * LP;
        #pragma unroll
        for (int s = L; s < LP; ++s) vr[s] = (_Float16)0.0f;
    }
    const v4f* qkv4 = (const v4f*)(qkv + (size_t)b * QKV_PER_WIN);
    for (int i4 = tid; i4 < QKV_PER_WIN / 4; i4 += 256) {
        const v4f val = qkv4[i4];
        #pragma unroll
        for (int j = 0; j < 4; ++j) {
            const int i  = i4 * 4 + j;
            const int ch = i / L;
            const int t  = i - ch * L;
            const int h  = ch / 96;
            const int r  = ch - h * 96;
            const float x = val[j];
            if (r < 32)      qT[(h * LP + t) * DK + r]        = (_Float16)(x * SCALE);
            else if (r < 64) kT[(h * LP + t) * DK + (r - 32)] = (_Float16)x;
            else             vS[(h * DK + (r - 64)) * LP + t] = (_Float16)x;
        }
    }
    const float* maskw = mask + w * (L * L);
    for (int i = tid; i < L * L; i += 256) { mS[i] = maskw[i]; iS[i] = rel_idx[i]; }
    for (int i = tid; i < 169 * NH; i += 256) bS[i] = bias_table[i];
    __syncthreads();

    // ---------- phase 2: S = qT^T * kT per head (16 WMMAs, K=32=d_k) ----------
    {
        const _Float16* qh = qT + wv * (LP * DK);
        const _Float16* kh = kT + wv * (LP * DK);
        float* Sh = S + wv * (L * LP);
        v16h bf[4];                                  // hoisted: invariant over mt
        #pragma unroll
        for (int nt = 0; nt < 4; ++nt)
            bf[nt] = load_b_frag(kh + (nt * 16 + lane16) * DK, 0, laneHi);
        #pragma unroll
        for (int mt = 0; mt < 4; ++mt) {
            const v16h a = load_a_frag(qh + (mt * 16 + lane16) * DK, 0, laneHi);
            v8f acc[4];
            #pragma unroll
            for (int nt = 0; nt < 4; ++nt) {         // 4 independent WMMAs back-to-back
                v8f z = {};
                acc[nt] = __builtin_amdgcn_wmma_f32_16x16x32_f16(
                            false, a, false, bf[nt], (short)0, z, false, false);
            }
            const int tb = mt * 16 + laneHi * 8;
            #pragma unroll
            for (int nt = 0; nt < 4; ++nt) {
                const int sI = nt * 16 + lane16;
                #pragma unroll
                for (int r2 = 0; r2 < 8; ++r2) {
                    const int t = tb + r2;
                    if (t < L) Sh[t * LP + sI] = acc[nt][r2];
                }
            }
        }
    }
    __syncthreads();

    // ---------- phase 3: + bias + mask, softmax over HEADS, P as f16 ----------
    // two s-columns per iteration: b64 S reads, packed v2h P writes.
    for (int p2 = tid; p2 < L * (LP / 2); p2 += 256) {
        const int t    = p2 >> 5;
        const int s0   = (p2 & 31) * 2;
        const int base = t * LP + s0;
        float x0[NH], x1[NH];
        #pragma unroll
        for (int h = 0; h < NH; ++h) {
            const v2f sv = *(const v2f*)(S + h * (L * LP) + base);
            x0[h] = sv[0]; x1[h] = sv[1];
        }
        const bool ok0 = (s0 < L);
        const bool ok1 = (s0 + 1 < L);
        if (ok0) { const int mi = t * L + s0;     head_softmax(x0, bS + iS[mi] * NH, mS[mi]); }
        if (ok1) { const int mi = t * L + s0 + 1; head_softmax(x1, bS + iS[mi] * NH, mS[mi]); }
        #pragma unroll
        for (int h = 0; h < NH; ++h) {
            v2h pr;
            pr[0] = ok0 ? (_Float16)x0[h] : (_Float16)0.0f;
            pr[1] = ok1 ? (_Float16)x1[h] : (_Float16)0.0f;   // pad cols -> 0 in P*V
            *(v2h*)(P + h * (LP * LP) + base) = pr;
        }
    }
    __syncthreads();

    // ---------- phase 4: out = V * P^T per head (16 WMMAs) ----------
    {
        const _Float16* vh = vS + wv * (DK * LP);
        const _Float16* Ph = P + wv * (LP * LP);
        float* outb = out + (size_t)b * (NH * DK * L) + wv * (DK * L);
        v16h av[2][2];                               // hoisted: invariant over tt
        #pragma unroll
        for (int ct = 0; ct < 2; ++ct)
            #pragma unroll
            for (int ks = 0; ks < 2; ++ks)
                av[ct][ks] = load_a_frag(vh + (ct * 16 + lane16) * LP, ks * 32, laneHi);
        #pragma unroll
        for (int tt = 0; tt < 4; ++tt) {
            const _Float16* pRow = Ph + (tt * 16 + lane16) * LP;
            const v16h b0 = load_b_frag(pRow, 0,  laneHi);
            const v16h b1 = load_b_frag(pRow, 32, laneHi);
            #pragma unroll
            for (int ct = 0; ct < 2; ++ct) {
                v8f acc = {};
                acc = __builtin_amdgcn_wmma_f32_16x16x32_f16(
                        false, av[ct][0], false, b0, (short)0, acc, false, false);
                acc = __builtin_amdgcn_wmma_f32_16x16x32_f16(
                        false, av[ct][1], false, b1, (short)0, acc, false, false);
                const int t = tt * 16 + lane16;
                if (t < L) {
                    const int cb = ct * 16 + laneHi * 8;
                    #pragma unroll
                    for (int r2 = 0; r2 < 8; ++r2)
                        outb[(cb + r2) * L + t] = acc[r2];
                }
            }
        }
    }
}

extern "C" void kernel_launch(void* const* d_in, const int* in_sizes, int n_in,
                              void* d_out, int out_size, void* d_ws, size_t ws_size,
                              hipStream_t stream) {
    const float* qkv        = (const float*)d_in[0];
    const float* mask       = (const float*)d_in[1];
    const float* bias_table = (const float*)d_in[2];
    const int*   rel_idx    = (const int*)d_in[3];
    float*       out        = (float*)d_out;

    const int nb = in_sizes[0] / QKV_PER_WIN;   // 4096 windows
    swin_window_attn_kernel<<<dim3(nb), dim3(256), SMEM_BYTES, stream>>>(
        qkv, mask, bias_table, rel_idx, out);
}
